// GGNN_14585708937254
// MI455X (gfx1250) — matrix operand order repaired and checked by reading, not compiled, not run on hardware
//
#include <hip/hip_runtime.h>
#include <hip/hip_bf16.h>
#include <math.h>

typedef __attribute__((ext_vector_type(16))) __bf16 v16bf;
typedef __attribute__((ext_vector_type(8)))  __bf16 v8bf;
typedef __attribute__((ext_vector_type(8)))  float  v8f;

#define CCH 128

__device__ __forceinline__ unsigned short f2bf(float f) {
    unsigned u = __float_as_uint(f);
    return (unsigned short)((u + 0x7FFFu + ((u >> 16) & 1u)) >> 16);
}

// ---------------- fp32 -> bf16 conversion (RNE) ----------------
__global__ __launch_bounds__(256)
void f32_to_bf16_k(const float* __restrict__ in, unsigned short* __restrict__ out, int n) {
    int i = blockIdx.x * 256 + threadIdx.x;
    if (i < n) out[i] = f2bf(in[i]);
}

// ---------------- fills ----------------
__global__ __launch_bounds__(256)
void fill_zero_k(float* __restrict__ p, int n) {
    int i = blockIdx.x * 256 + threadIdx.x;
    if (i < n) p[i] = 0.0f;
}
__global__ __launch_bounds__(256)
void zero_int_k(int* __restrict__ p, int n) {
    int i = blockIdx.x * 256 + threadIdx.x;
    if (i < n) p[i] = 0;
}

// ---------------- weight pack into WMMA B-fragment layout ----------------
// Packed per matrix: idx = ((tileN*4 + kc)*32 + lane)*16 + e
//   col n = tileN*16 + (lane&15);  k = kc*32 + (lane>>4)*16 + e
__global__ __launch_bounds__(256)
void pack_b_k(const float* __restrict__ W, unsigned short* __restrict__ P,
              int Ncols, int nMat, int transposed) {
    int per = Ncols * 128;
    int idx = blockIdx.x * 256 + threadIdx.x;
    if (idx >= per * nMat) return;
    int m = idx / per;
    int r = idx % per;
    int e    = r & 15;
    int lane = (r >> 4) & 31;
    int kc   = (r >> 9) & 3;
    int tN   = r >> 11;
    int n = tN * 16 + (lane & 15);
    int k = kc * 32 + (lane >> 4) * 16 + e;
    float v = transposed ? W[(size_t)m * per + (size_t)n * 128 + k]
                         : W[(size_t)m * per + (size_t)k * Ncols + n];
    P[idx] = f2bf(v);
}

// ---------------- conv GEMM: strip-mined, B staged in LDS ----------------
__global__ __launch_bounds__(256)
void gemm_conv_k(const __bf16* __restrict__ A, const __bf16* __restrict__ Bp,
                 float* __restrict__ C, int M) {
    __shared__ unsigned short bsm[16384];   // 128x128 bf16 packed = 32KB
    {
        const uint4* s = (const uint4*)Bp;
        uint4* d = (uint4*)bsm;
#pragma unroll
        for (int i = 0; i < 8; ++i)
            d[threadIdx.x + i * 256] = s[threadIdx.x + i * 256];
    }
    __syncthreads();
    int wave = threadIdx.x >> 5, lane = threadIdx.x & 31;
    int mT = blockIdx.x * 8 + wave;
    if (mT * 16 >= M) return;
    int l16 = lane & 15, half = lane >> 4;

    const __bf16* arow = A + (size_t)(mT * 16 + l16) * 128 + half * 16;
    v16bf a[4];
#pragma unroll
    for (int kc = 0; kc < 4; ++kc) {
        v8bf alo = *(const v8bf*)(arow + kc * 32);
        v8bf ahi = *(const v8bf*)(arow + kc * 32 + 8);
#pragma unroll
        for (int j = 0; j < 8; ++j) { a[kc][j] = alo[j]; a[kc][8 + j] = ahi[j]; }
    }
    const __bf16* bl = (const __bf16*)bsm;
#pragma unroll
    for (int nt = 0; nt < 8; ++nt) {
        v8f acc = {};
#pragma unroll
        for (int kc = 0; kc < 4; ++kc) {
            v16bf b = *(const v16bf*)(bl + ((size_t)(nt * 4 + kc) * 32 + lane) * 16);
            acc = __builtin_amdgcn_wmma_f32_16x16x32_bf16(
                      false, a[kc], false, b, (short)0, acc, false, false);
        }
        float* crow = C + (size_t)(mT * 16 + half * 8) * 128 + nt * 16 + l16;
#pragma unroll
        for (int j = 0; j < 8; ++j) crow[(size_t)j * 128] = acc[j];
    }
}

// ---------------- CSR build ----------------
__global__ __launch_bounds__(256)
void count_k(const int* __restrict__ dst, int* __restrict__ counts, int E) {
    int e = blockIdx.x * 256 + threadIdx.x;
    if (e < E) atomicAdd(&counts[dst[e]], 1);
}
__global__ __launch_bounds__(256)
void chunk_sum_k(const int* __restrict__ counts, int* __restrict__ chunksum, int N) {
    __shared__ int s[256];
    int i = blockIdx.x * 256 + threadIdx.x;
    s[threadIdx.x] = (i < N) ? counts[i] : 0;
    __syncthreads();
    for (int d = 128; d > 0; d >>= 1) {
        if (threadIdx.x < d) s[threadIdx.x] += s[threadIdx.x + d];
        __syncthreads();
    }
    if (threadIdx.x == 0) chunksum[blockIdx.x] = s[0];
}
__global__ void scan_chunks_k(const int* __restrict__ chunksum,
                              int* __restrict__ chunkoff, int nChunks) {
    if (threadIdx.x == 0 && blockIdx.x == 0) {
        int run = 0;
        for (int i = 0; i < nChunks; ++i) { chunkoff[i] = run; run += chunksum[i]; }
    }
}
__global__ __launch_bounds__(256)
void scan_within_k(const int* __restrict__ counts, const int* __restrict__ chunkoff,
                   int* __restrict__ rowstart, int N) {
    __shared__ int s[256];
    int i = blockIdx.x * 256 + threadIdx.x;
    int v = (i < N) ? counts[i] : 0;
    s[threadIdx.x] = v;
    __syncthreads();
    for (int d = 1; d < 256; d <<= 1) {
        int t = (threadIdx.x >= d) ? s[threadIdx.x - d] : 0;
        __syncthreads();
        s[threadIdx.x] += t;
        __syncthreads();
    }
    if (i < N) rowstart[i] = chunkoff[blockIdx.x] + s[threadIdx.x] - v;
}
__global__ __launch_bounds__(256)
void fill_csr_k(const int* __restrict__ src, const int* __restrict__ dst,
                const float* __restrict__ ew, int* __restrict__ cursor,
                int* __restrict__ csr_src, float* __restrict__ csr_w, int E) {
    int e = blockIdx.x * 256 + threadIdx.x;
    if (e >= E) return;
    int pos = atomicAdd(&cursor[dst[e]], 1);
    csr_src[pos] = src[e];
    csr_w[pos] = ew[e];
}

// ---------------- gather aggregation: one wave per dst node, no atomics -------
__global__ __launch_bounds__(256)
void aggregate_k(const float* __restrict__ m, const int* __restrict__ rowstart,
                 const int* __restrict__ counts, const int* __restrict__ csr_src,
                 const float* __restrict__ csr_w, unsigned short* __restrict__ aggb,
                 int N, int useEw) {
    int wv = blockIdx.x * 8 + (threadIdx.x >> 5);
    if (wv >= N) return;
    int lane = threadIdx.x & 31;
    int cb = lane * 4;
    int beg = rowstart[wv], cnt = counts[wv];
    float a0 = 0.f, a1 = 0.f, a2 = 0.f, a3 = 0.f;
    for (int j = 0; j < cnt; ++j) {
        int s = csr_src[beg + j];
        if (j + 1 < cnt)
            __builtin_prefetch(m + (size_t)csr_src[beg + j + 1] * CCH + cb, 0, 3);
        float w = useEw ? csr_w[beg + j] : 1.0f;
        const float4 v = *(const float4*)(m + (size_t)s * CCH + cb);
        a0 += w * v.x; a1 += w * v.y; a2 += w * v.z; a3 += w * v.w;
    }
    uint2 u;
    u.x = (unsigned)f2bf(a0) | ((unsigned)f2bf(a1) << 16);
    u.y = (unsigned)f2bf(a2) | ((unsigned)f2bf(a3) << 16);
    *(uint2*)(aggb + (size_t)wv * CCH + cb) = u;
}

// ---------------- fused GRU: gi/gh GEMMs + gates, LDS-staged B ----------------
// wave -> 16-row strip. Per column group t, the whole block needs only 6 B tiles
// ({t,t+8,t+16} from wih and whh) = 24KB: stage them in LDS cooperatively, then
// 24 WMMAs from ds_load, gates in registers, write h (f32) + hb (bf16).
__global__ __launch_bounds__(256)
void gru_fused_k(const __bf16* __restrict__ Aagg, const __bf16* __restrict__ Ah,
                 const __bf16* __restrict__ BpI, const __bf16* __restrict__ BpH,
                 const float* __restrict__ bih, const float* __restrict__ bhh,
                 float* __restrict__ h, unsigned short* __restrict__ hb, int N) {
    __shared__ unsigned short bsm[6 * 2048];   // 6 tiles x 4KB = 24KB
    int wave = threadIdx.x >> 5, lane = threadIdx.x & 31;
    int mT = blockIdx.x * 8 + wave;
    bool valid = (mT * 16 < N);
    int mTs = valid ? mT : 0;
    int l16 = lane & 15, half = lane >> 4;

    const __bf16* ar  = Aagg + (size_t)(mTs * 16 + l16) * 128 + half * 16;
    const __bf16* ahp = Ah   + (size_t)(mTs * 16 + l16) * 128 + half * 16;
    v16bf aa[4], ahf[4];
#pragma unroll
    for (int kc = 0; kc < 4; ++kc) {
        v8bf lo  = *(const v8bf*)(ar  + kc * 32);
        v8bf hi  = *(const v8bf*)(ar  + kc * 32 + 8);
        v8bf lo2 = *(const v8bf*)(ahp + kc * 32);
        v8bf hi2 = *(const v8bf*)(ahp + kc * 32 + 8);
#pragma unroll
        for (int j = 0; j < 8; ++j) {
            aa[kc][j]  = lo[j];  aa[kc][8 + j]  = hi[j];
            ahf[kc][j] = lo2[j]; ahf[kc][8 + j] = hi2[j];
        }
    }
    for (int t = 0; t < 8; ++t) {
        // cooperative stage: q=0..2 -> BpI tiles {t,t+8,t+16}; q=3..5 -> BpH
        {
            uint4* d = (uint4*)bsm;
#pragma unroll
            for (int q = 0; q < 6; ++q) {
                int tile = (q % 3 == 0) ? t : (q % 3 == 1) ? (t + 8) : (t + 16);
                const uint4* sp = (q < 3) ? (const uint4*)BpI : (const uint4*)BpH;
                d[q * 256 + threadIdx.x] = sp[(size_t)tile * 256 + threadIdx.x];
            }
        }
        __syncthreads();
        if (valid) {
            v8f ir = {}, iz = {}, in_ = {}, hrv = {}, hz = {}, hn = {};
            const __bf16* bl = (const __bf16*)bsm;
#pragma unroll
            for (int kc = 0; kc < 4; ++kc) {
                size_t fo = (size_t)kc * 512 + (size_t)lane * 16;
                ir  = __builtin_amdgcn_wmma_f32_16x16x32_bf16(false, aa[kc],  false, *(const v16bf*)(bl + 0 * 2048 + fo), (short)0, ir,  false, false);
                iz  = __builtin_amdgcn_wmma_f32_16x16x32_bf16(false, aa[kc],  false, *(const v16bf*)(bl + 1 * 2048 + fo), (short)0, iz,  false, false);
                in_ = __builtin_amdgcn_wmma_f32_16x16x32_bf16(false, aa[kc],  false, *(const v16bf*)(bl + 2 * 2048 + fo), (short)0, in_, false, false);
                hrv = __builtin_amdgcn_wmma_f32_16x16x32_bf16(false, ahf[kc], false, *(const v16bf*)(bl + 3 * 2048 + fo), (short)0, hrv, false, false);
                hz  = __builtin_amdgcn_wmma_f32_16x16x32_bf16(false, ahf[kc], false, *(const v16bf*)(bl + 4 * 2048 + fo), (short)0, hz,  false, false);
                hn  = __builtin_amdgcn_wmma_f32_16x16x32_bf16(false, ahf[kc], false, *(const v16bf*)(bl + 5 * 2048 + fo), (short)0, hn,  false, false);
            }
            int c = t * 16 + l16;
            float br = bih[c], bz = bih[128 + c], bn = bih[256 + c];
            float cr = bhh[c], cz = bhh[128 + c], cn = bhh[256 + c];
            float* hcol = h + (size_t)(mT * 16 + half * 8) * 128 + c;
            unsigned short* hbcol = hb + (size_t)(mT * 16 + half * 8) * 128 + c;
#pragma unroll
            for (int j = 0; j < 8; ++j) {
                float hv = hcol[(size_t)j * 128];
                float r = 1.0f / (1.0f + __expf(-(ir[j] + br + hrv[j] + cr)));
                float z = 1.0f / (1.0f + __expf(-(iz[j] + bz + hz[j] + cz)));
                float nn = tanhf(in_[j] + bn + r * (hn[j] + cn));
                float o = (1.0f - z) * nn + z * hv;
                hcol[(size_t)j * 128] = o;
                hbcol[(size_t)j * 128] = f2bf(o);
            }
        }
        __syncthreads();
    }
}

// ---------------- GraphNorm stats (optionally fused leaky-relu) ---------------
__global__ __launch_bounds__(256)
void gn_stats_k(float* __restrict__ h, float* __restrict__ sums, int N, int doLrelu) {
    __shared__ float s0[256];
    __shared__ float s1[256];
    int c    = threadIdx.x & 127;
    int half = threadIdx.x >> 7;
    int r0 = blockIdx.x * 64;
    float s = 0.0f, q = 0.0f;
    for (int r = half; r < 64; r += 2) {
        int row = r0 + r;
        if (row < N) {
            float v = h[(size_t)row * CCH + c];
            if (doLrelu) {
                v = v > 0.0f ? v : 0.01f * v;
                h[(size_t)row * CCH + c] = v;
            }
            s += v; q += v * v;
        }
    }
    s0[threadIdx.x] = s; s1[threadIdx.x] = q;
    __syncthreads();
    if (half == 0) {
        s += s0[threadIdx.x + 128];
        q += s1[threadIdx.x + 128];
        atomicAdd(&sums[c], s);
        atomicAdd(&sums[128 + c], q);
    }
}

// ---------------- GraphNorm apply (dual output, optional post-lrelu) ----------
__global__ __launch_bounds__(256)
void gn_apply_k(float* __restrict__ h, unsigned short* __restrict__ hb,
                const float* __restrict__ sums, const float* __restrict__ w,
                const float* __restrict__ b, const float* __restrict__ ms,
                int N, int postLrelu) {
    int idx = blockIdx.x * 256 + threadIdx.x;
    if (idx >= N * CCH) return;
    int c = idx & 127;
    float invN = 1.0f / (float)N;
    float mean = sums[c] * invN;
    float ex2  = sums[128 + c] * invN;
    float m = ms[c];
    float var = ex2 - mean * mean * (2.0f * m - m * m);
    float xc = h[idx] - m * mean;
    float v = w[c] * xc * rsqrtf(var + 1e-5f) + b[c];
    if (postLrelu) v = v > 0.0f ? v : 0.01f * v;
    h[idx] = v; hb[idx] = f2bf(v);
}

// ---------------- final: lrelu + segment_sum over batch ----------------
__global__ __launch_bounds__(256)
void final_sum_k(const float* __restrict__ h, const int* __restrict__ batch,
                 float* __restrict__ out, int N) {
    int idx = blockIdx.x * 256 + threadIdx.x;
    if (idx >= N * CCH) return;
    int n = idx >> 7;
    int c = idx & 127;
    float v = h[idx];
    v = v > 0.0f ? v : 0.01f * v;
    atomicAdd(&out[(size_t)batch[n] * CCH + c], v);
}

extern "C" void kernel_launch(void* const* d_in, const int* in_sizes, int n_in,
                              void* d_out, int out_size, void* d_ws, size_t ws_size,
                              hipStream_t stream) {
    (void)n_in; (void)ws_size;
    const float* x     = (const float*)d_in[0];
    const float* ew    = (const float*)d_in[1];
    const float* convw = (const float*)d_in[2];
    const float* wih   = (const float*)d_in[3];
    const float* whh   = (const float*)d_in[4];
    const float* bih   = (const float*)d_in[5];
    const float* bhh   = (const float*)d_in[6];
    const float* gnw   = (const float*)d_in[7];
    const float* gnb   = (const float*)d_in[8];
    const float* gnms  = (const float*)d_in[9];
    const int*   eidx  = (const int*)d_in[10];
    const int*   batch = (const int*)d_in[11];
    float* out = (float*)d_out;

    const int N  = in_sizes[0] / CCH;   // 40000
    const int E  = in_sizes[1];         // 640000
    const int NC = N * CCH;
    const int* src = eidx;
    const int* dst = eidx + E;

    // ---- workspace layout ----
    char* ws = (char*)d_ws;
    size_t off = 0;
    auto alloc = [&](size_t bytes) -> char* {
        char* p = ws + off;
        off += (bytes + 255) & ~(size_t)255;
        return p;
    };
    float* h     = (float*)alloc((size_t)NC * 4);
    float* mbuf  = (float*)alloc((size_t)NC * 4);
    float* sums  = (float*)alloc(256 * 4);
    int* counts   = (int*)alloc((size_t)N * 4);
    int* rowstart = (int*)alloc((size_t)N * 4);
    int* cursor   = (int*)alloc((size_t)N * 4);
    int* chunksum = (int*)alloc(4096);
    int* chunkoff = (int*)alloc(4096);
    int* csr_src  = (int*)alloc((size_t)E * 4);
    float* csr_w  = (float*)alloc((size_t)E * 4);
    unsigned short* hb    = (unsigned short*)alloc((size_t)NC * 2);
    unsigned short* aggb  = (unsigned short*)alloc((size_t)NC * 2);
    unsigned short* convP = (unsigned short*)alloc((size_t)25 * 16384 * 2);
    unsigned short* wihP  = (unsigned short*)alloc((size_t)5 * 49152 * 2);
    unsigned short* whhP  = (unsigned short*)alloc((size_t)5 * 49152 * 2);

    // ---- pack weights (bf16 WMMA fragment layout) ----
    {
        int tot = 25 * 16384;
        pack_b_k<<<(tot + 255) / 256, 256, 0, stream>>>(convw, convP, 128, 25, 0);
        tot = 5 * 49152;
        pack_b_k<<<(tot + 255) / 256, 256, 0, stream>>>(wih, wihP, 384, 5, 1);
        pack_b_k<<<(tot + 255) / 256, 256, 0, stream>>>(whh, whhP, 384, 5, 1);
    }

    // ---- CSR build (once per launch) ----
    const int nChunks = (N + 255) / 256;
    zero_int_k<<<(N + 255) / 256, 256, 0, stream>>>(counts, N);
    count_k<<<(E + 255) / 256, 256, 0, stream>>>(dst, counts, E);
    chunk_sum_k<<<nChunks, 256, 0, stream>>>(counts, chunksum, N);
    scan_chunks_k<<<1, 64, 0, stream>>>(chunksum, chunkoff, nChunks);
    scan_within_k<<<nChunks, 256, 0, stream>>>(counts, chunkoff, rowstart, N);
    hipMemcpyAsync(cursor, rowstart, (size_t)N * 4, hipMemcpyDeviceToDevice, stream);
    fill_csr_k<<<(E + 255) / 256, 256, 0, stream>>>(src, dst, ew, cursor, csr_src, csr_w, E);

    // ---- h = x (f32 + bf16 mirror) ----
    hipMemcpyAsync(h, x, (size_t)NC * 4, hipMemcpyDeviceToDevice, stream);
    f32_to_bf16_k<<<(NC + 255) / 256, 256, 0, stream>>>(x, hb, NC);

    const int steps[5] = {5, 4, 3, 2, 1};
    const int ncBlocks = (NC + 255) / 256;
    const int mTiles = (N + 15) / 16;                 // 2500
    const int tileBlocks = (mTiles + 7) / 8;          // 313
    const int nodeBlocks = (N + 7) / 8;
    const int statBlocks = (N + 63) / 64;

    for (int L = 0; L < 5; ++L) {
        for (int s = 0; s < steps[L]; ++s) {
            gemm_conv_k<<<tileBlocks, 256, 0, stream>>>(
                (const __bf16*)hb,
                (const __bf16*)(convP + (size_t)(L * 5 + s) * 16384), mbuf, N);
            aggregate_k<<<nodeBlocks, 256, 0, stream>>>(
                mbuf, rowstart, counts, csr_src, csr_w, aggb, N, (L < 4) ? 1 : 0);
            gru_fused_k<<<tileBlocks, 256, 0, stream>>>(
                (const __bf16*)aggb, (const __bf16*)hb,
                (const __bf16*)(wihP + (size_t)L * 49152),
                (const __bf16*)(whhP + (size_t)L * 49152),
                bih + L * 384, bhh + L * 384, h, hb, N);
        }
        if (L == 0) {
            // graph_norm then lrelu (stats on raw h, lrelu fused into apply)
            fill_zero_k<<<1, 256, 0, stream>>>(sums, 256);
            gn_stats_k<<<statBlocks, 256, 0, stream>>>(h, sums, N, 0);
            gn_apply_k<<<ncBlocks, 256, 0, stream>>>(h, hb, sums, gnw, gnb, gnms, N, 1);
        } else if (L < 4) {
            // lrelu then graph_norm (lrelu fused into the stats pass)
            fill_zero_k<<<1, 256, 0, stream>>>(sums, 256);
            gn_stats_k<<<statBlocks, 256, 0, stream>>>(h, sums, N, 1);
            gn_apply_k<<<ncBlocks, 256, 0, stream>>>(
                h, hb, sums, gnw + L * 128, gnb + L * 128, gnms + L * 128, N, 0);
        } else {
            // lrelu fused into the final segment sum
            fill_zero_k<<<(out_size + 255) / 256, 256, 0, stream>>>(out, out_size);
            final_sum_k<<<ncBlocks, 256, 0, stream>>>(h, batch, out, N);
        }
    }
}
